// SCConv_77455440216408
// MI455X (gfx1250) — matrix operand order repaired and checked by reading, not compiled, not run on hardware
//
#include <hip/hip_runtime.h>
#include <stdint.h>

// ---------------------------------------------------------------------------
// SCConv forward for MI455X (gfx1250, wave32, WMMA).
//  - Neighborhood matrices converted to bf16 once (plus transposes, ad1+au1).
//  - All GEMMs via v_wmma_f32_16x16x32_bf16, fp32 accumulation.
//  - Aggregation GEMMs: split-K (4 deterministic slices -> partial buffers),
//    B panel staged in LDS per workgroup via async global->LDS double
//    buffering (s_wait_asynccnt), A streamed from HBM once w/ prefetch.
//  - Fused slice-reduction + sigmoid re-quantization to bf16.
// Workspace: ~342 MB.
// ---------------------------------------------------------------------------

typedef __attribute__((ext_vector_type(16))) __bf16 v16bf;
typedef __attribute__((ext_vector_type(8)))  __bf16 v8bf;
typedef __attribute__((ext_vector_type(8)))  float  v8f;
typedef int v4i32 __attribute__((vector_size(16)));   // matches async-LDS builtin

union V16U { v16bf v; v8bf h[2]; };

#define KC 64                 // k-chunk staged in LDS
#define KCP (KC + 8)          // padded row stride (144B): conflict-free b128
#define SLICES 4              // split-K factor

#if defined(__gfx1250__) && __has_builtin(__builtin_amdgcn_global_load_async_to_lds_b128)
#define HAVE_ASYNC_LDS 1
#else
#define HAVE_ASYNC_LDS 0
#endif

__device__ __forceinline__ unsigned short f2bf(float f) {
  union { float f; unsigned int u; } c; c.f = f;
  unsigned int r = c.u + 0x7fffu + ((c.u >> 16) & 1u);   // round-nearest-even
  return (unsigned short)(r >> 16);
}
__device__ __forceinline__ float bf2f(unsigned short b) {
  union { unsigned int u; float f; } c; c.u = ((unsigned int)b) << 16;
  return c.f;
}

// ---------------------------------------------------------------------------
// Elementwise kernels
// ---------------------------------------------------------------------------
__global__ void cvt_bf16(unsigned short* __restrict__ dst,
                         const float* __restrict__ src, int n) {
  int i = blockIdx.x * blockDim.x + threadIdx.x;
  if (i < n) dst[i] = f2bf(src[i]);
}

__global__ void add_cvt_bf16(unsigned short* __restrict__ dst,
                             const float* __restrict__ a,
                             const float* __restrict__ b, int n) {
  int i = blockIdx.x * blockDim.x + threadIdx.x;
  if (i < n) dst[i] = f2bf(a[i] + b[i]);
}

// xb[i] = bf16( sigmoid( sum_p parts[p*n + i] ) )
__global__ void reduce_sigmoid(const float* __restrict__ parts, int nparts,
                               int n, unsigned short* __restrict__ xb) {
  int i = blockIdx.x * blockDim.x + threadIdx.x;
  if (i >= n) return;
  float s = 0.f;
  for (int p = 0; p < nparts; ++p) s += parts[(size_t)p * n + i];
  xb[i] = f2bf(1.0f / (1.0f + __expf(-s)));
}

// Transpose R x Ccols fp32 -> Ccols x R bf16 (R, Ccols multiples of 32).
__global__ __launch_bounds__(256) void transpose_cvt(
    unsigned short* __restrict__ dst, const float* __restrict__ src,
    int R, int Ccols) {
  __shared__ float tile[32][33];
  const int gc = blockIdx.x * 32, gr = blockIdx.y * 32;
  #pragma unroll
  for (int i = 0; i < 4; ++i)
    tile[threadIdx.y + i * 8][threadIdx.x] =
        src[(size_t)(gr + threadIdx.y + i * 8) * Ccols + gc + threadIdx.x];
  __syncthreads();
  #pragma unroll
  for (int i = 0; i < 4; ++i)
    dst[(size_t)(gc + threadIdx.y + i * 8) * R + gr + threadIdx.x] =
        f2bf(tile[threadIdx.x][threadIdx.y + i * 8]);
}

// ---------------------------------------------------------------------------
// Feature transform: Tt(128 x M, bf16) = (X(M x 128, bf16) @ W(128x128, f32))^T
// Computed as W^T @ X^T with WMMA; W^T staged in LDS as bf16.
// Block = 256 threads (8 waves), tile = 128(n) x 64(m) of Tt, grid = M/64.
// ---------------------------------------------------------------------------
__global__ __launch_bounds__(256) void feat_transform(
    const unsigned short* __restrict__ X,   // M x 128 bf16 row-major
    const float* __restrict__ W,            // 128 x 128 f32 row-major
    unsigned short* __restrict__ Tt,        // 128 x M bf16 row-major
    int M) {
  __shared__ __align__(32) unsigned short WT[128 * 128];  // WT[n*128+k]=W[k,n]
  for (int i = threadIdx.x; i < 128 * 128; i += 256) {
    int k = i >> 7, n = i & 127;
    WT[n * 128 + k] = f2bf(W[i]);
  }
  __syncthreads();

  const int wave = threadIdx.x >> 5;
  const int lane = threadIdx.x & 31;
  const int half = lane >> 4;
  const int l16  = lane & 15;
  const int n0   = wave * 16;
  const int m0   = blockIdx.x * 64;

  v8f acc[4] = {};
  for (int k0 = 0; k0 < 128; k0 += 32) {
    V16U a;   // 16-bit A layout from LDS
    const unsigned short* ar = WT + (size_t)(n0 + l16) * 128 + k0;
    a.h[0] = *(const v8bf*)(ar + half * 16);
    a.h[1] = *(const v8bf*)(ar + half * 16 + 8);
    #pragma unroll
    for (int mt = 0; mt < 4; ++mt) {
      const v16bf b = *(const v16bf*)(
          X + (size_t)(m0 + mt * 16 + l16) * 128 + k0 + half * 16);
      acc[mt] = __builtin_amdgcn_wmma_f32_16x16x32_bf16(
          false, a.v, false, b, (short)0, acc[mt], false, false);
    }
  }
  #pragma unroll
  for (int mt = 0; mt < 4; ++mt) {
    #pragma unroll
    for (int v = 0; v < 8; ++v) {
      int n = n0 + v + half * 8;
      int m = m0 + mt * 16 + l16;
      Tt[(size_t)n * M + m] = f2bf(acc[mt][v]);
    }
  }
}

// ---------------------------------------------------------------------------
// B-panel staging: copy 128 x KC bf16 chunk of Bt into LDS (padded rows).
// 128 threads x 8 chunks of 16B each. Async (gfx1250) or plain fallback.
// ---------------------------------------------------------------------------
__device__ __forceinline__ void stage_B(unsigned short* __restrict__ dst,
                                        const unsigned short* __restrict__ Bt,
                                        int K, int k0, int tid) {
  #pragma unroll
  for (int i = 0; i < 8; ++i) {
    const int c    = tid + i * 128;   // 0..1023
    const int n    = c >> 3;          // 0..127
    const int koff = (c & 7) * 8;     // 0,8,..,56
#if HAVE_ASYNC_LDS
    __builtin_amdgcn_global_load_async_to_lds_b128(
        (__attribute__((address_space(1))) v4i32*)(Bt + (size_t)n * K + k0 + koff),
        (__attribute__((address_space(3))) v4i32*)(dst + n * KCP + koff),
        0, 0);
#else
    *(v8bf*)(dst + n * KCP + koff) =
        *(const v8bf*)(Bt + (size_t)n * K + k0 + koff);
#endif
  }
}

__device__ __forceinline__ void wait_stage(bool pending_next) {
#if HAVE_ASYNC_LDS
 #if __has_builtin(__builtin_amdgcn_s_wait_asynccnt)
  if (pending_next) __builtin_amdgcn_s_wait_asynccnt(8);
  else              __builtin_amdgcn_s_wait_asynccnt(0);
 #else
  if (pending_next) asm volatile("s_wait_asynccnt 0x8" ::: "memory");
  else              asm volatile("s_wait_asynccnt 0x0" ::: "memory");
 #endif
#endif
}

// ---------------------------------------------------------------------------
// Aggregation GEMM, split-K:
//   Cout[s] (M x 128, f32) = A(M x K/4 slice, bf16) @ B(K/4 x 128, bf16)
// B supplied pre-transposed as Bt(128 x K). Block = 128 threads (4 waves),
// each wave owns a 16 x 128 output strip. grid = (M/64, SLICES).
// B chunks double-buffered in LDS via async global->LDS.
// ---------------------------------------------------------------------------
__global__ __launch_bounds__(128) void agg_gemm(
    const unsigned short* __restrict__ A,
    const unsigned short* __restrict__ Bt,
    float* __restrict__ Cout,      // SLICES consecutive M x 128 buffers
    int M, int K) {
  __shared__ __align__(32) unsigned short bs[2][128 * KCP];

  const int tid  = threadIdx.x;
  const int wave = tid >> 5;
  const int lane = tid & 31;
  const int half = lane >> 4;
  const int l16  = lane & 15;
  const int m0   = blockIdx.x * 64 + wave * 16;
  const int kbeg = blockIdx.y * (K >> 2);
  const int kend = kbeg + (K >> 2);
  float* __restrict__ Cs = Cout + (size_t)blockIdx.y * M * 128;

  v8f acc[8] = {};
  const unsigned short* arow = A + (size_t)(m0 + l16) * K;

  stage_B(&bs[0][0], Bt, K, kbeg, tid);
  int buf = 0;
  for (int k0 = kbeg; k0 < kend; k0 += KC, buf ^= 1) {
    const bool has_next = (k0 + KC) < kend;
    if (has_next) stage_B(&bs[buf ^ 1][0], Bt, K, k0 + KC, tid);
    wait_stage(has_next);
    __syncthreads();

    #pragma unroll
    for (int kk = 0; kk < KC; kk += 32) {
      V16U a;  // contiguous k order (same as B): lane<16 -> K {0..15}; lane>=16 -> {16..31}
      a.h[0] = *(const v8bf*)(arow + k0 + kk + half * 16);
      a.h[1] = *(const v8bf*)(arow + k0 + kk + half * 16 + 8);
      __builtin_prefetch(arow + k0 + kk + 1024, 0, 1);  // global_prefetch_b8
      #pragma unroll
      for (int nt = 0; nt < 8; ++nt) {
        const v16bf b = *(const v16bf*)&bs[buf][(nt * 16 + l16) * KCP + kk + half * 16];
        acc[nt] = __builtin_amdgcn_wmma_f32_16x16x32_bf16(
            false, a.v, false, b, (short)0, acc[nt], false, false);
      }
    }
    __syncthreads();  // all waves done reading bs[buf] before it is restaged
  }

  #pragma unroll
  for (int nt = 0; nt < 8; ++nt) {
    #pragma unroll
    for (int v = 0; v < 8; ++v) {
      size_t idx = (size_t)(m0 + v + half * 8) * 128 + nt * 16 + l16;
      Cs[idx] = acc[nt][v];
    }
  }
}

// ---------------------------------------------------------------------------
// Head: out[c] = sum_r ( mean_rows(x_r) . lin_r_w[:,c] + lin_r_b[c] )
// ---------------------------------------------------------------------------
__global__ __launch_bounds__(128) void head_kernel(
    const unsigned short* __restrict__ x0, const unsigned short* __restrict__ x1,
    const unsigned short* __restrict__ x2,
    const float* __restrict__ w0, const float* __restrict__ b0,
    const float* __restrict__ w1, const float* __restrict__ b1,
    const float* __restrict__ w2, const float* __restrict__ b2,
    float* __restrict__ out) {
  __shared__ float m0s[128], m1s[128], m2s[128];
  const int j = threadIdx.x;
  float s0 = 0.f, s1 = 0.f, s2 = 0.f;
  for (int i = 0; i < 2048; ++i) s0 += bf2f(x0[(size_t)i * 128 + j]);
  for (int i = 0; i < 6144; ++i) s1 += bf2f(x1[(size_t)i * 128 + j]);
  for (int i = 0; i < 4096; ++i) s2 += bf2f(x2[(size_t)i * 128 + j]);
  m0s[j] = s0 / 2048.f; m1s[j] = s1 / 6144.f; m2s[j] = s2 / 4096.f;
  __syncthreads();
  if (j < 2) {
    float r = b0[j] + b1[j] + b2[j];
    for (int k = 0; k < 128; ++k)
      r += m0s[k] * w0[k * 2 + j] + m1s[k] * w1[k * 2 + j] + m2s[k] * w2[k * 2 + j];
    out[j] = r;
  }
}

// ---------------------------------------------------------------------------
// Launcher
// ---------------------------------------------------------------------------
extern "C" void kernel_launch(void* const* d_in, const int* in_sizes, int n_in,
                              void* d_out, int out_size, void* d_ws, size_t ws_size,
                              hipStream_t stream) {
  constexpr int NN = 2048, NE = 6144, NF = 4096, C = 128;

  const float* x0f   = (const float*)d_in[0];
  const float* x1f   = (const float*)d_in[1];
  const float* x2f   = (const float*)d_in[2];
  const float* inc1  = (const float*)d_in[3];
  const float* inc1n = (const float*)d_in[4];
  const float* inc2  = (const float*)d_in[5];
  const float* inc2n = (const float*)d_in[6];
  const float* au0   = (const float*)d_in[7];
  const float* au1   = (const float*)d_in[8];
  const float* ad1   = (const float*)d_in[9];
  const float* ad2   = (const float*)d_in[10];
  const float* W00   = (const float*)d_in[11];
  const float* W10   = (const float*)d_in[12];
  const float* W01   = (const float*)d_in[13];
  const float* W11   = (const float*)d_in[14];
  const float* W21   = (const float*)d_in[15];
  const float* W12   = (const float*)d_in[16];
  const float* W22   = (const float*)d_in[17];
  const float* l0w = (const float*)d_in[18]; const float* l0b = (const float*)d_in[19];
  const float* l1w = (const float*)d_in[20]; const float* l1b = (const float*)d_in[21];
  const float* l2w = (const float*)d_in[22]; const float* l2b = (const float*)d_in[23];

  unsigned char* ws = (unsigned char*)d_ws;
  size_t off = 0;
  auto alloc = [&](size_t bytes) -> void* {
    void* p = ws + off;
    off += (bytes + 255) & ~(size_t)255;
    return p;
  };

  // bf16 neighborhood operands (built once; constant across layers)
  unsigned short* au0b    = (unsigned short*)alloc((size_t)NN * NN * 2);
  unsigned short* inc1nb  = (unsigned short*)alloc((size_t)NN * NE * 2);
  unsigned short* inc1Tb  = (unsigned short*)alloc((size_t)NE * NN * 2);
  unsigned short* adsum1b = (unsigned short*)alloc((size_t)NE * NE * 2);
  unsigned short* inc2nb  = (unsigned short*)alloc((size_t)NE * NF * 2);
  unsigned short* inc2Tb  = (unsigned short*)alloc((size_t)NF * NE * 2);
  unsigned short* ad2b    = (unsigned short*)alloc((size_t)NF * NF * 2);
  // rank features (bf16)
  unsigned short* xb0 = (unsigned short*)alloc((size_t)NN * C * 2);
  unsigned short* xb1 = (unsigned short*)alloc((size_t)NE * C * 2);
  unsigned short* xb2 = (unsigned short*)alloc((size_t)NF * C * 2);
  // transposed transformed features
  unsigned short* t00T = (unsigned short*)alloc((size_t)C * NN * 2);
  unsigned short* t10T = (unsigned short*)alloc((size_t)C * NE * 2);
  unsigned short* t01T = (unsigned short*)alloc((size_t)C * NN * 2);
  unsigned short* t11T = (unsigned short*)alloc((size_t)C * NE * 2);
  unsigned short* t21T = (unsigned short*)alloc((size_t)C * NF * 2);
  unsigned short* t12T = (unsigned short*)alloc((size_t)C * NE * 2);
  unsigned short* t22T = (unsigned short*)alloc((size_t)C * NF * 2);
  // split-K partials: rank0: 2 gemms, rank1: 3 gemms, rank2: 2 gemms
  float* part0 = (float*)alloc((size_t)2 * SLICES * NN * C * 4);
  float* part1 = (float*)alloc((size_t)3 * SLICES * NE * C * 4);
  float* part2 = (float*)alloc((size_t)2 * SLICES * NF * C * 4);

  auto cvt = [&](unsigned short* d, const float* s, int n) {
    cvt_bf16<<<(n + 255) / 256, 256, 0, stream>>>(d, s, n);
  };

  // ---- one-time operand preparation ----
  cvt(au0b,   au0,   NN * NN);
  cvt(inc1nb, inc1n, NN * NE);
  cvt(inc2nb, inc2n, NE * NF);
  cvt(ad2b,   ad2,   NF * NF);
  add_cvt_bf16<<<((NE * NE) + 255) / 256, 256, 0, stream>>>(adsum1b, ad1, au1, NE * NE);
  transpose_cvt<<<dim3(NE / 32, NN / 32), dim3(32, 8), 0, stream>>>(inc1Tb, inc1, NN, NE);
  transpose_cvt<<<dim3(NF / 32, NE / 32), dim3(32, 8), 0, stream>>>(inc2Tb, inc2, NE, NF);
  cvt(xb0, x0f, NN * C);
  cvt(xb1, x1f, NE * C);
  cvt(xb2, x2f, NF * C);

  // ---- layers ----
  for (int l = 0; l < 2; ++l) {
    const size_t wo = (size_t)l * C * C;
    feat_transform<<<NN / 64, 256, 0, stream>>>(xb0, W00 + wo, t00T, NN);
    feat_transform<<<NE / 64, 256, 0, stream>>>(xb1, W10 + wo, t10T, NE);
    feat_transform<<<NN / 64, 256, 0, stream>>>(xb0, W01 + wo, t01T, NN);
    feat_transform<<<NE / 64, 256, 0, stream>>>(xb1, W11 + wo, t11T, NE);
    feat_transform<<<NF / 64, 256, 0, stream>>>(xb2, W21 + wo, t21T, NF);
    feat_transform<<<NE / 64, 256, 0, stream>>>(xb1, W12 + wo, t12T, NE);
    feat_transform<<<NF / 64, 256, 0, stream>>>(xb2, W22 + wo, t22T, NF);

    agg_gemm<<<dim3(NN / 64, SLICES), 128, 0, stream>>>(au0b,    t00T,
        part0 + (size_t)0 * SLICES * NN * C, NN, NN);
    agg_gemm<<<dim3(NN / 64, SLICES), 128, 0, stream>>>(inc1nb,  t10T,
        part0 + (size_t)1 * SLICES * NN * C, NN, NE);
    agg_gemm<<<dim3(NE / 64, SLICES), 128, 0, stream>>>(inc1Tb,  t01T,
        part1 + (size_t)0 * SLICES * NE * C, NE, NN);
    agg_gemm<<<dim3(NE / 64, SLICES), 128, 0, stream>>>(adsum1b, t11T,
        part1 + (size_t)1 * SLICES * NE * C, NE, NE);
    agg_gemm<<<dim3(NE / 64, SLICES), 128, 0, stream>>>(inc2nb,  t21T,
        part1 + (size_t)2 * SLICES * NE * C, NE, NF);
    agg_gemm<<<dim3(NF / 64, SLICES), 128, 0, stream>>>(inc2Tb,  t12T,
        part2 + (size_t)0 * SLICES * NF * C, NF, NE);
    agg_gemm<<<dim3(NF / 64, SLICES), 128, 0, stream>>>(ad2b,    t22T,
        part2 + (size_t)1 * SLICES * NF * C, NF, NF);

    reduce_sigmoid<<<(NN * C + 255) / 256, 256, 0, stream>>>(part0, 2 * SLICES, NN * C, xb0);
    reduce_sigmoid<<<(NE * C + 255) / 256, 256, 0, stream>>>(part1, 3 * SLICES, NE * C, xb1);
    reduce_sigmoid<<<(NF * C + 255) / 256, 256, 0, stream>>>(part2, 2 * SLICES, NF * C, xb2);
  }

  head_kernel<<<1, 128, 0, stream>>>(xb0, xb1, xb2, l0w, l0b, l1w, l1b, l2w, l2b,
                                     (float*)d_out);
  (void)in_sizes; (void)n_in; (void)out_size; (void)ws_size;
}
